// MultiHeadSelfAttention_44865228374217
// MI455X (gfx1250) — compile-verified, matrix operand order checked by reading, not run on hardware
//
#include <hip/hip_runtime.h>
#include <math.h>

#define DMODEL 2048
#define SEQ    2048
#define NH     16
#define DH     128
#define NROWS  4096  // B*S

typedef __attribute__((ext_vector_type(16))) __bf16 v16bf;
typedef __attribute__((ext_vector_type(8)))  float  v8f;

union FragA { v16bf v; unsigned u[8]; };

__device__ __forceinline__ v8f wmma_bf16(v16bf a, v16bf b, v8f c) {
#if defined(__HIP_DEVICE_COMPILE__)
  return __builtin_amdgcn_wmma_f32_16x16x32_bf16(false, a, false, b, (short)0, c,
                                                 false, false);
#else
  return c;
#endif
}

// K-offset of bf16 pair v (0..7) for the 16-bit A-matrix lane layout.
// lane half 0 holds K=[0..7]+16*(v/4), half 1 holds K=[8..15]+16*(v/4).
__device__ __forceinline__ int kbase_of(int vv, int half) {
  return ((vv >> 2) << 4) + (half << 3) + ((vv & 3) << 1);
}

// Async 16B global->LDS copy (CDNA5 VGLOBAL async op, tracked by ASYNCcnt).
// lds_off: byte offset within the wave's LDS (low 32 bits of flat shared addr).
__device__ __forceinline__ void async_ld16(unsigned lds_off, const void* g) {
  asm volatile("global_load_async_to_lds_b128 %0, %1, off"
               :: "v"(lds_off), "v"(g) : "memory");
}
#define ASYNC_WAIT_8() asm volatile("s_wait_asynccnt 0x8" ::: "memory")
#define ASYNC_WAIT_0() asm volatile("s_wait_asynccnt 0x0" ::: "memory")

// ---------------------------------------------------------------------------
// 128x128 tile GEMM: out = A[4096x2048] @ W[2048x2048] + bias
// MODE 0: write bf16*(1/sqrt(Dh)) to outB in [B,H,S,Dh] layout (Q, ws)
// MODE 1: f32 -> outF [B,H,S,Dh] (K output)  + bf16 copy -> outB2 same layout
// MODE 2: f32 -> outF [B,H,S,Dh] (V output)  + bf16 transposed -> outB2 [B,H,Dh,S]
// MODE 3: f32 -> outF row-major [B*S, D] (attn_out)
// ---------------------------------------------------------------------------
template <bool A_BF16, int MODE>
__global__ __launch_bounds__(256)
void gemm128(const void* __restrict__ Aptr, const float* __restrict__ W,
             const float* __restrict__ bias, float* __restrict__ outF,
             __bf16* __restrict__ outB, __bf16* __restrict__ outB2) {
  __shared__ __bf16 As[128 * 36];  // [row][k], padded stride
  __shared__ __bf16 Bs[128 * 36];  // [n][k] == W^T tile

  const int t = threadIdx.x;
  const int wv = t >> 5, lane = t & 31, l16 = lane & 15, half = lane >> 4;
  const int r0 = blockIdx.x * 128, n0 = blockIdx.y * 128;

  v8f acc[8];
  for (int i = 0; i < 8; ++i)
    for (int j = 0; j < 8; ++j) acc[i][j] = 0.f;

  const int ac = t & 31, ar = t >> 5;    // A staging: k-col, row-base
  const int bn = t & 127, bk0 = t >> 7;  // B staging: n, k parity
  const float*  Af = (const float*)Aptr;
  const __bf16* Ab = (const __bf16*)Aptr;

  for (int kt = 0; kt < DMODEL / 32; ++kt) {
    const int kk = kt * 32;
    __syncthreads();
    for (int i = 0; i < 16; ++i) {
      int r = ar + (i << 3);
      __bf16 hv;
      if (A_BF16) hv = Ab[(size_t)(r0 + r) * DMODEL + kk + ac];
      else        hv = (__bf16)Af[(size_t)(r0 + r) * DMODEL + kk + ac];
      As[r * 36 + ac] = hv;
    }
    for (int j = 0; j < 16; ++j) {
      int kl = bk0 + (j << 1);
      Bs[bn * 36 + kl] = (__bf16)W[(size_t)(kk + kl) * DMODEL + n0 + bn];
    }
    __syncthreads();

    FragA a;
    const int am = wv * 16 + l16;
    for (int vv = 0; vv < 8; ++vv)
      a.u[vv] = *(const unsigned*)&As[am * 36 + kbase_of(vv, half)];
    for (int nt = 0; nt < 8; ++nt) {
      FragA b;
      const int bnn = nt * 16 + l16;
      for (int vv = 0; vv < 8; ++vv)
        b.u[vv] = *(const unsigned*)&Bs[bnn * 36 + kbase_of(vv, half)];
      acc[nt] = wmma_bf16(a.v, b.v, acc[nt]);
    }
  }

  for (int nt = 0; nt < 8; ++nt) {
    for (int r = 0; r < 8; ++r) {
      int row = r0 + wv * 16 + (half << 3) + r;  // C layout: M = 8*half + r
      int col = n0 + nt * 16 + l16;
      float vv = acc[nt][r] + bias[col];
      if (MODE == 3) {
        outF[(size_t)row * DMODEL + col] = vv;
      } else {
        int b = row >> 11, s = row & (SEQ - 1);
        int h = col >> 7, dd = col & (DH - 1);
        size_t idx = (((size_t)(b * NH + h)) * SEQ + s) * DH + dd;
        if (MODE == 0) {
          outB[idx] = (__bf16)(vv * 0.08838834764831845f);  // fold 1/sqrt(Dh)
        } else {
          outF[idx] = vv;
          if (MODE == 1) outB2[idx] = (__bf16)vv;
          else  /* V^T */ outB2[(((size_t)(b * NH + h)) * DH + dd) * SEQ + s] =
                              (__bf16)vv;
        }
      }
    }
  }
}

// ---------------------------------------------------------------------------
// Flash attention: one block per (b,h, 128-query tile), key blocks of 64,
// causal masking, online softmax, bf16 WMMA for QK^T and P*V.
// K/V tiles stream through double-buffered LDS via GLOBAL_LOAD_ASYNC_TO_LDS.
// q: bf16 [B,H,S,Dh] pre-scaled (ws); kb16: bf16 [B,H,S,Dh] (ws);
// vt16: bf16 [B,H,Dh,S] (ws); ctx: bf16 [B,S,D] (ws).
// ---------------------------------------------------------------------------
__global__ __launch_bounds__(256)
void attn_flash(const __bf16* __restrict__ q, const __bf16* __restrict__ kb16,
                const __bf16* __restrict__ vt16, __bf16* __restrict__ ctx) {
  __shared__ __bf16 Ks[2][64 * 136];   // [key][Dh], 272B rows (16B aligned)
  __shared__ __bf16 Vs[2][128 * 72];   // [Dh][key], 144B rows (16B aligned)
  __shared__ __bf16 Ps[8][16 * 68];    // per-wave P scratch [row][key]

  const int t = threadIdx.x;
  const int wv = t >> 5, lane = t & 31, l16 = lane & 15, half = lane >> 4;
  const int qt = blockIdx.x & 15;
  const int bh = blockIdx.x >> 4;
  const int b = bh >> 4, h = bh & 15;

  // Preload Q fragments (Dh=128 -> 4 WMMA K-steps), A-layout row = lane%16.
  const int qrow_lane = qt * 128 + wv * 16 + l16;
  FragA qf[4];
  {
    const __bf16* qp = q + ((size_t)bh * SEQ + qrow_lane) * DH;
    for (int ks = 0; ks < 4; ++ks)
      for (int vv = 0; vv < 8; ++vv)
        qf[ks].u[vv] = *(const unsigned*)(qp + ks * 32 + kbase_of(vv, half));
  }

  v8f O[8];
  for (int i = 0; i < 8; ++i)
    for (int j = 0; j < 8; ++j) O[i][j] = 0.f;
  float mst[8], lst[8];
  for (int r = 0; r < 8; ++r) { mst[r] = -__builtin_inff(); lst[r] = 0.f; }

  const int nkb = 2 * qt + 2;  // causal: keys up to (qt*128 + 127)
  const __bf16* Kbase = kb16 + (size_t)bh * SEQ * DH;
  const __bf16* Vbase = vt16 + (size_t)bh * DH * SEQ;

  // Issue one key block's async staging: 8 x b128 per wave (4 K + 4 V).
  auto issue = [&](int buf, int kb) {
    const int k0 = kb * 64;
    for (int j = 0; j < 4; ++j) {           // K: 64 rows x 256B
      int c = t + (j << 8);
      int key = c >> 4, xc = (c & 15) << 3;
      async_ld16((unsigned)(size_t)&Ks[buf][key * 136 + xc],
                 Kbase + ((size_t)(k0 + key) * DH + xc));
    }
    for (int j = 0; j < 4; ++j) {           // V^T: 128 rows x 128B
      int c = t + (j << 8);
      int d = c >> 3, kc = (c & 7) << 3;
      async_ld16((unsigned)(size_t)&Vs[buf][d * 72 + kc],
                 Vbase + ((size_t)d * SEQ + k0 + kc));
    }
  };

  issue(0, 0);
  for (int kb = 0; kb < nkb; ++kb) {
    const int k0 = kb * 64;
    __syncthreads();  // prev compute done before overwriting buf[(kb+1)&1]
    if (kb + 1 < nkb) { issue((kb + 1) & 1, kb + 1); ASYNC_WAIT_8(); }
    else              { ASYNC_WAIT_0(); }
    __syncthreads();  // current buffer visible to all waves
    const __bf16* Kc = Ks[kb & 1];
    const __bf16* Vc = Vs[kb & 1];

    // scores = Q @ K^T for 64 keys (scale pre-folded into Q)
    v8f sc[4];
    for (int i = 0; i < 4; ++i)
      for (int j = 0; j < 8; ++j) sc[i][j] = 0.f;
    for (int ks = 0; ks < 4; ++ks) {
      for (int nt = 0; nt < 4; ++nt) {
        FragA bfr;
        int n = nt * 16 + l16;
        for (int vv = 0; vv < 8; ++vv)
          bfr.u[vv] = *(const unsigned*)&Kc[n * 136 + ks * 32 + kbase_of(vv, half)];
        sc[nt] = wmma_bf16(qf[ks].v, bfr.v, sc[nt]);
      }
    }

    // causal mask
    for (int nt = 0; nt < 4; ++nt) {
      int kcol = k0 + nt * 16 + l16;
      for (int r = 0; r < 8; ++r) {
        int qrow = qt * 128 + wv * 16 + (half << 3) + r;
        sc[nt][r] = (kcol <= qrow) ? sc[nt][r] : -__builtin_inff();
      }
    }
    // online softmax: row stats via width-16 butterflies (C layout: n = lane%16)
    float mnew[8], alpha[8];
    for (int r = 0; r < 8; ++r) {
      float mx = fmaxf(fmaxf(sc[0][r], sc[1][r]), fmaxf(sc[2][r], sc[3][r]));
      for (int d = 1; d < 16; d <<= 1) mx = fmaxf(mx, __shfl_xor(mx, d, 16));
      mnew[r] = fmaxf(mst[r], mx);
      alpha[r] = __expf(mst[r] - mnew[r]);
      mst[r] = mnew[r];
    }
    float rs[8];
    for (int r = 0; r < 8; ++r) rs[r] = 0.f;
    for (int nt = 0; nt < 4; ++nt)
      for (int r = 0; r < 8; ++r) {
        float p = __expf(sc[nt][r] - mnew[r]);
        sc[nt][r] = p;
        rs[r] += p;
      }
    for (int r = 0; r < 8; ++r) {
      float s = rs[r];
      for (int d = 1; d < 16; d <<= 1) s += __shfl_xor(s, d, 16);
      lst[r] = lst[r] * alpha[r] + s;
      for (int nt = 0; nt < 8; ++nt) O[nt][r] *= alpha[r];
    }

    // P: C layout -> A layout via per-wave LDS scratch
    for (int nt = 0; nt < 4; ++nt)
      for (int r = 0; r < 8; ++r)
        Ps[wv][((half << 3) + r) * 68 + nt * 16 + l16] = (__bf16)sc[nt][r];
    asm volatile("s_wait_dscnt 0x0" ::: "memory");

    // O += P @ V (64-key reduction = 2 WMMA K-steps)
    for (int ks = 0; ks < 2; ++ks) {
      FragA pa;
      for (int vv = 0; vv < 8; ++vv)
        pa.u[vv] = *(const unsigned*)&Ps[wv][l16 * 68 + ks * 32 + kbase_of(vv, half)];
      for (int nt = 0; nt < 8; ++nt) {
        FragA vb;
        int n = nt * 16 + l16;
        for (int vv = 0; vv < 8; ++vv)
          vb.u[vv] = *(const unsigned*)&Vc[n * 72 + ks * 32 + kbase_of(vv, half)];
        O[nt] = wmma_bf16(pa.v, vb.v, O[nt]);
      }
    }
  }

  // normalize, write ctx back in [B,S,D] layout (heads re-interleaved)
  for (int r = 0; r < 8; ++r) {
    float inv = 1.0f / lst[r];
    int qrow = qt * 128 + wv * 16 + (half << 3) + r;
    for (int nt = 0; nt < 8; ++nt) {
      int col = h * DH + nt * 16 + l16;
      ctx[((size_t)b * SEQ + qrow) * DMODEL + col] = (__bf16)(O[nt][r] * inv);
    }
  }
}

// ---------------------------------------------------------------------------
// y = attn_out + x; ln = gamma * (y - mu) * rsqrt(var + eps) + beta
// ---------------------------------------------------------------------------
__global__ __launch_bounds__(256)
void ln_residual(const float* __restrict__ x, const float* __restrict__ attn,
                 const float* __restrict__ gamma, const float* __restrict__ beta,
                 float* __restrict__ ln) {
  __shared__ float s1[8], s2[8];
  const int row = blockIdx.x;
  const float* xr = x + (size_t)row * DMODEL;
  const float* ar = attn + (size_t)row * DMODEL;
  float y[8], sum = 0.f, sq = 0.f;
  for (int i = 0; i < 8; ++i) {
    int j = threadIdx.x + (i << 8);
    float yv = xr[j] + ar[j];
    y[i] = yv; sum += yv; sq += yv * yv;
  }
  for (int d = 1; d < 32; d <<= 1) {
    sum += __shfl_xor(sum, d, 32);
    sq  += __shfl_xor(sq, d, 32);
  }
  int wv = threadIdx.x >> 5;
  if ((threadIdx.x & 31) == 0) { s1[wv] = sum; s2[wv] = sq; }
  __syncthreads();
  float ts = 0.f, tq = 0.f;
  for (int i = 0; i < 8; ++i) { ts += s1[i]; tq += s2[i]; }
  float mu   = ts * (1.0f / DMODEL);
  float var  = tq * (1.0f / DMODEL) - mu * mu;
  float rstd = rsqrtf(var + 1e-5f);
  for (int i = 0; i < 8; ++i) {
    int j = threadIdx.x + (i << 8);
    ln[(size_t)row * DMODEL + j] = gamma[j] * (y[i] - mu) * rstd + beta[j];
  }
}

extern "C" void kernel_launch(void* const* d_in, const int* in_sizes, int n_in,
                              void* d_out, int out_size, void* d_ws, size_t ws_size,
                              hipStream_t stream) {
  const float* x     = (const float*)d_in[0];
  const float* Wq    = (const float*)d_in[1];
  const float* bq    = (const float*)d_in[2];
  const float* Wk    = (const float*)d_in[3];
  const float* bk    = (const float*)d_in[4];
  const float* Wv    = (const float*)d_in[5];
  const float* bv    = (const float*)d_in[6];
  const float* Wo    = (const float*)d_in[7];
  const float* bo    = (const float*)d_in[8];
  const float* gamma = (const float*)d_in[9];
  const float* beta  = (const float*)d_in[10];
  (void)in_sizes; (void)n_in; (void)out_size; (void)ws_size;

  const size_t N = (size_t)NROWS * DMODEL;  // 8388608 per output tensor
  float* out     = (float*)d_out;
  float* outLN   = out;
  float* outAttn = out + N;
  float* outK    = out + 2 * N;
  float* outV    = out + 3 * N;

  __bf16* qws   = (__bf16*)d_ws;                                      // 16 MB
  __bf16* ctxws = (__bf16*)((char*)d_ws + 1 * N * sizeof(__bf16));    // 16 MB
  __bf16* kbws  = (__bf16*)((char*)d_ws + 2 * N * sizeof(__bf16));    // 16 MB
  __bf16* vtws  = (__bf16*)((char*)d_ws + 3 * N * sizeof(__bf16));    // 16 MB

  dim3 blk(256);
  dim3 gg(NROWS / 128, DMODEL / 128);

  gemm128<false, 0><<<gg, blk, 0, stream>>>(x, Wq, bq, (float*)nullptr, qws,
                                            (__bf16*)nullptr);
  gemm128<false, 1><<<gg, blk, 0, stream>>>(x, Wk, bk, outK, (__bf16*)nullptr,
                                            kbws);
  gemm128<false, 2><<<gg, blk, 0, stream>>>(x, Wv, bv, outV, (__bf16*)nullptr,
                                            vtws);

  attn_flash<<<dim3(2 * NH * (SEQ / 128)), blk, 0, stream>>>(qws, kbws, vtws,
                                                             ctxws);

  gemm128<true, 3><<<gg, blk, 0, stream>>>(ctxws, Wo, bo, outAttn,
                                           (__bf16*)nullptr, (__bf16*)nullptr);

  ln_residual<<<dim3(NROWS), blk, 0, stream>>>(x, outAttn, gamma, beta, outLN);
}
